// FeatureAlign_1717986918860
// MI455X (gfx1250) — matrix operand order repaired and checked by reading, not compiled, not run on hardware
//
#include <hip/hip_runtime.h>
#include <hip/hip_bf16.h>

// ---------------------------------------------------------------------------
// FeatureAlign (DCNv2 alignment block) for gfx1250 / MI455X.
// All GEMM-shaped work runs on V_WMMA_F32_16X16X32_BF16 (fp32 accumulate).
// Weights pre-converted to bf16 (com_w also tap-major transposed) so the
// WMMA inner loops are pure b128 loads + wmma, no fp32->bf16 conversion.
// ---------------------------------------------------------------------------

typedef __bf16 bf16_t;
typedef __attribute__((ext_vector_type(16))) __bf16 v16bf;
typedef __attribute__((ext_vector_type(8)))  __bf16 v8bf;
typedef __attribute__((ext_vector_type(8)))  float  v8f;

#define B_    2
#define C_    128
#define H_    128
#define W_    128
#define HW_   (H_ * W_)
#define HS_   64
#define WS_   64
#define HWS_  (HS_ * WS_)
#define DG_   8
#define KK_   9
#define CG_   16
#define COUT_ 64
#define OMC_  216          // 3 * DG * KK
#define KDCN_ 1152         // C * 9 == DG*CG*KK

__device__ __forceinline__ v8f wmma_bf16(v16bf a, v16bf b, v8f c) {
  return __builtin_amdgcn_wmma_f32_16x16x32_bf16(
      /*neg_a=*/false, a, /*neg_b=*/false, b,
      /*c_mod=*/(short)0, c, /*reuse_a=*/false, /*reuse_b=*/false);
}

__device__ __forceinline__ v16bf combine16(v8bf lo, v8bf hi) {
  return __builtin_shufflevector(lo, hi, 0, 1, 2, 3, 4, 5, 6, 7,
                                 8, 9, 10, 11, 12, 13, 14, 15);
}

// A fragment (16x32 bf16): row = lane%16; half 0 holds K {k0..k0+7,k0+16..k0+23},
// half 1 holds K {k0+8..k0+15,k0+24..k0+31}.  Two aligned 16B loads of bf16.
__device__ __forceinline__ v16bf load_a_bf16(const bf16_t* __restrict__ w,
                                             int row, int ldk, int k0, int half) {
  const bf16_t* p = w + (size_t)row * ldk + k0 + half * 8;
  return combine16(*(const v8bf*)p, *(const v8bf*)(p + 16));
}

// B fragment (32x16 bf16): col = lane%16; half 0 holds K {k0..k0+7,k0+16..k0+23},
// half 1 holds K {k0+8..k0+15,k0+24..k0+31} (matches A).
// Source: LDS column vector (channels contiguous per pixel), two b128 ds loads.
__device__ __forceinline__ v16bf load_b_lds(const bf16_t* __restrict__ col,
                                            int k0, int half) {
  const bf16_t* p = col + k0 + half * 8;
  return combine16(*(const v8bf*)p, *(const v8bf*)(p + 16));
}

// ---------------------------------------------------------------------------
// Kernel 0: one-shot weight conversion fp32 -> bf16; com_w -> [tap][oc][c].
// ---------------------------------------------------------------------------
__global__ __launch_bounds__(256) void k0_cvt_weights(
    const float* __restrict__ fsm_w, const float* __restrict__ offset_w,
    const float* __restrict__ com_w, const float* __restrict__ dcn_w,
    const float* __restrict__ cat_w,
    bf16_t* __restrict__ fsm_b, bf16_t* __restrict__ offw_b,
    bf16_t* __restrict__ comt_b, bf16_t* __restrict__ dcnw_b,
    bf16_t* __restrict__ catw_b) {
  const int i0 = blockIdx.x * 256 + threadIdx.x;
  const int stride = gridDim.x * 256;
  for (int t = i0; t < C_ * C_; t += stride) fsm_b[t] = (bf16_t)fsm_w[t];
  for (int t = i0; t < C_ * 2 * C_; t += stride) offw_b[t] = (bf16_t)offset_w[t];
  for (int t = i0; t < COUT_ * KDCN_; t += stride) dcnw_b[t] = (bf16_t)dcn_w[t];
  for (int t = i0; t < C_ * COUT_; t += stride) catw_b[t] = (bf16_t)cat_w[t];
  // comt[((tap*216)+oc)*128 + c] = com_w[oc*1152 + c*9 + tap]
  for (int t = i0; t < 9 * OMC_ * C_; t += stride) {
    int tap = t / (OMC_ * C_);
    int rem = t - tap * (OMC_ * C_);
    int oc = rem / C_, c = rem - (rem / C_) * C_;
    comt_b[t] = (bf16_t)com_w[(size_t)oc * KDCN_ + c * 9 + tap];
  }
}

// ---------------------------------------------------------------------------
// Kernel 1: feat_arm = relu(fsm_w @ feat_l)   (M=128,K=128)
//           off_feat = relu(offset_w @ [feat_arm ; 2*feat_up])  (M=128,K=256)
// One 16-pixel tile per block, 8 waves (one 16-row M tile each).
// off_feat stored bf16 (consumed only as WMMA operand in k2).
// ---------------------------------------------------------------------------
__global__ __launch_bounds__(256) void k1_fsm_offset(
    const float* __restrict__ feat_l, const float* __restrict__ feat_s,
    const bf16_t* __restrict__ fsm_b, const bf16_t* __restrict__ offw_b,
    float* __restrict__ arm_g, bf16_t* __restrict__ off_bf) {
  __shared__ __align__(16) bf16_t Xs[16][C_ + 8];       // feat_l tile [n][c]
  __shared__ __align__(16) bf16_t Cat[16][2 * C_ + 8];  // [arm ; 2*feat_up]

  const int tid  = threadIdx.x;
  const int lane = tid & 31, wave = tid >> 5;
  const int half = lane >> 4, ln = lane & 15;

  const int p0  = blockIdx.x * 16;
  const int b   = p0 / HW_;
  const int hw0 = p0 % HW_;
  const int h   = hw0 / W_;
  const int w0  = hw0 % W_;

  for (int i = tid; i < C_ * 16; i += 256) {
    int c = i >> 4, n = i & 15;
    Xs[n][c] = (bf16_t)feat_l[(size_t)(b * C_ + c) * HW_ + hw0 + n];
    // nearest-neighbor upsample: feat_up[h][w] = feat_s[h>>1][w>>1]
    float up = feat_s[(size_t)(b * C_ + c) * HWS_ + (h >> 1) * WS_ + ((w0 + n) >> 1)];
    Cat[n][C_ + c] = (bf16_t)(2.0f * up);
  }
  __syncthreads();

  { // Stage 1: FSM 1x1 conv + relu
    const int m0 = wave * 16;
    v8f acc = {};
#pragma unroll
    for (int k0 = 0; k0 < C_; k0 += 32) {
      v16bf a  = load_a_bf16(fsm_b, m0 + ln, C_, k0, half);
      v16bf bb = load_b_lds(&Xs[ln][0], k0, half);
      acc = wmma_bf16(a, bb, acc);
    }
#pragma unroll
    for (int r = 0; r < 8; ++r) {
      int row = m0 + half * 8 + r;
      float v = acc[r] > 0.f ? acc[r] : 0.f;
      arm_g[(size_t)(b * C_ + row) * HW_ + hw0 + ln] = v;
      Cat[ln][row] = (bf16_t)v;
    }
  }
  __syncthreads();

  { // Stage 2: offset 1x1 conv (K=256) + relu
    const int m0 = wave * 16;
    v8f acc = {};
#pragma unroll
    for (int k0 = 0; k0 < 2 * C_; k0 += 32) {
      v16bf a  = load_a_bf16(offw_b, m0 + ln, 2 * C_, k0, half);
      v16bf bb = load_b_lds(&Cat[ln][0], k0, half);
      acc = wmma_bf16(a, bb, acc);
    }
#pragma unroll
    for (int r = 0; r < 8; ++r) {
      int row = m0 + half * 8 + r;
      float v = acc[r] > 0.f ? acc[r] : 0.f;
      off_bf[(size_t)(b * C_ + row) * HW_ + hw0 + ln] = (bf16_t)v;
    }
  }
}

// ---------------------------------------------------------------------------
// Kernel 2: om = conv3x3(off_feat, com_w) + com_b  (M=216, K=1152, im2col).
// 9 tap-GEMMs of K=128 accumulate into one fragment; 3x18 bf16 halo in LDS.
// grid.y=2 halves of M, 7 waves/block, rows clamped at 216.
// ---------------------------------------------------------------------------
__global__ __launch_bounds__(224) void k2_com_conv(
    const bf16_t* __restrict__ off_bf, const bf16_t* __restrict__ comt_b,
    const float* __restrict__ com_b, float* __restrict__ om_g) {
  __shared__ __align__(16) bf16_t Bs[3][C_][20];  // [dy][ch][w0-1..w0+16], 0-pad

  const int tid  = threadIdx.x;
  const int lane = tid & 31, wave = tid >> 5;
  const int half = lane >> 4, ln = lane & 15;

  const int p0  = blockIdx.x * 16;
  const int b   = p0 / HW_;
  const int hw0 = p0 % HW_;
  const int h   = hw0 / W_;
  const int w0  = hw0 % W_;

  for (int i = tid; i < 3 * C_ * 18; i += 224) {
    int dy  = i / (C_ * 18);
    int rem = i - dy * (C_ * 18);
    int c   = rem / 18;
    int x   = rem - c * 18;
    int yy = h + dy - 1;
    int xx = w0 + x - 1;
    bf16_t v = (bf16_t)0.f;
    if (yy >= 0 && yy < H_ && xx >= 0 && xx < W_)
      v = off_bf[(size_t)(b * C_ + c) * HW_ + yy * W_ + xx];
    Bs[dy][c][x] = v;
  }
  __syncthreads();

  const int m0   = blockIdx.y * 112 + wave * 16;
  const int row  = m0 + ln;
  const int rowc = row < OMC_ ? row : OMC_ - 1;
  v8f acc = {};
#pragma unroll
  for (int tap = 0; tap < 9; ++tap) {
    const int ky = tap / 3, kx = tap % 3;
    const bf16_t* wt = comt_b + ((size_t)tap * OMC_ + rowc) * C_;  // contiguous c
#pragma unroll
    for (int k0 = 0; k0 < C_; k0 += 32) {
      const bf16_t* p = wt + k0 + half * 8;
      v16bf a = combine16(*(const v8bf*)p, *(const v8bf*)(p + 16));
      v16bf bb;
#pragma unroll
      for (int j = 0; j < 16; ++j) bb[j] = Bs[ky][k0 + half * 8 + (j & 7) + (j >> 3) * 16][ln + kx];
      acc = wmma_bf16(a, bb, acc);
    }
  }
#pragma unroll
  for (int r = 0; r < 8; ++r) {
    int orow = m0 + half * 8 + r;
    if (orow < OMC_)
      om_g[(size_t)(b * OMC_ + orow) * HW_ + hw0 + ln] = acc[r] + com_b[orow];
  }
}

// ---------------------------------------------------------------------------
// Kernel 3: deformable bilinear sampling -> val tile (bf16, im2col-K order),
// DCN contraction (M=64,K=1152) + bias + relu, cat 1x1 conv (M=128,K=64)
// + relu + residual add of feat_arm.  4 waves / 16-pixel tile.
// ---------------------------------------------------------------------------
__global__ __launch_bounds__(128) void k3_dcn(
    const float* __restrict__ feat_s, const float* __restrict__ om_g,
    const bf16_t* __restrict__ dcnw_b, const float* __restrict__ dcn_b,
    const bf16_t* __restrict__ catw_b, const float* __restrict__ arm_g,
    float* __restrict__ out_g) {
  __shared__ __align__(16) bf16_t Vs[16][KDCN_ + 8];  // [pixel][(g*16+c)*9+kk]
  __shared__ __align__(16) bf16_t Fa[16][COUT_ + 8];  // feat_align tile

  const int tid  = threadIdx.x;
  const int lane = tid & 31, wave = tid >> 5;
  const int half = lane >> 4, ln = lane & 15;

  const int p0  = blockIdx.x * 16;
  const int b   = p0 / HW_;
  const int hw0 = p0 % HW_;
  const int h   = hw0 / W_;
  const int w0  = hw0 % W_;

  // Phase 1: bilinear gather * sigmoid(mask). 1152 (n,g,kk) triples, 9/thread.
  for (int t = tid; t < 16 * DG_ * KK_; t += 128) {
    int n  = t / (DG_ * KK_);
    int r  = t - n * (DG_ * KK_);
    int g  = r / KK_;
    int kk = r - g * KK_;
    int hw = hw0 + n;
    // concat([o1,o2]) channel l == om channel l; mask channel = 144 + g*9 + kk
    float oy = om_g[(size_t)(b * OMC_ + (g * 18 + kk * 2)) * HW_ + hw];
    float ox = om_g[(size_t)(b * OMC_ + (g * 18 + kk * 2 + 1)) * HW_ + hw];
    float mz = om_g[(size_t)(b * OMC_ + (144 + g * KK_ + kk)) * HW_ + hw];
    float mask = 1.f / (1.f + __expf(-mz));

    float py = (float)h + (float)(kk / 3 - 1) + oy;
    float px = (float)(w0 + n) + (float)(kk % 3 - 1) + ox;
    float fy = floorf(py), fx = floorf(px);
    float ly = py - fy, lx = px - fx;
    int y0 = (int)fy, x0 = (int)fx;

    int   yi[2] = {y0, y0 + 1};
    int   xi[2] = {x0, x0 + 1};
    float wy[2] = {1.f - ly, ly};
    float wx[2] = {1.f - lx, lx};
    int cidx[4]; float cw[4];
#pragma unroll
    for (int cy = 0; cy < 2; ++cy)
#pragma unroll
      for (int cx = 0; cx < 2; ++cx) {
        int q = cy * 2 + cx;
        bool ok = (yi[cy] >= 0) && (yi[cy] < H_) && (xi[cx] >= 0) && (xi[cx] < W_);
        int yc = yi[cy] < 0 ? 0 : (yi[cy] > H_ - 1 ? H_ - 1 : yi[cy]);
        int xc = xi[cx] < 0 ? 0 : (xi[cx] > W_ - 1 ? W_ - 1 : xi[cx]);
        cidx[q] = (yc >> 1) * WS_ + (xc >> 1);  // feat_up == nearest(feat_s)
        cw[q]   = ok ? wy[cy] * wx[cx] * mask : 0.f;
      }
    const float* ch0 = feat_s + (size_t)(b * C_ + g * CG_) * HWS_;
#pragma unroll
    for (int c = 0; c < CG_; ++c) {
      const float* ch = ch0 + (size_t)c * HWS_;
      float v = cw[0] * ch[cidx[0]] + cw[1] * ch[cidx[1]] +
                cw[2] * ch[cidx[2]] + cw[3] * ch[cidx[3]];
      Vs[n][(g * CG_ + c) * KK_ + kk] = (bf16_t)v;
    }
  }
  __syncthreads();

  { // Phase 2: DCN einsum as GEMM, M=64 over 4 waves, K=1152
    const int m0 = wave * 16;
    v8f acc = {};
    for (int k0 = 0; k0 < KDCN_; k0 += 32) {
      if (k0 + 128 < KDCN_)  // stream next A chunk (global_prefetch_b8)
        __builtin_prefetch((const void*)(dcnw_b + (size_t)(m0 + ln) * KDCN_ +
                                         k0 + 128), 0, 1);
      v16bf a  = load_a_bf16(dcnw_b, m0 + ln, KDCN_, k0, half);
      v16bf bb = load_b_lds(&Vs[ln][0], k0, half);
      acc = wmma_bf16(a, bb, acc);
    }
#pragma unroll
    for (int r = 0; r < 8; ++r) {
      int row = m0 + half * 8 + r;
      float v = acc[r] + dcn_b[row];
      Fa[ln][row] = (bf16_t)(v > 0.f ? v : 0.f);
    }
  }
  __syncthreads();

  // Phase 3: final = relu(cat_w @ feat_align) + feat_arm   (M=128, K=64)
#pragma unroll
  for (int mb = 0; mb < 2; ++mb) {
    const int m0 = wave * 16 + mb * 64;
    v8f acc = {};
#pragma unroll
    for (int k0 = 0; k0 < COUT_; k0 += 32) {
      v16bf a  = load_a_bf16(catw_b, m0 + ln, COUT_, k0, half);
      v16bf bb = load_b_lds(&Fa[ln][0], k0, half);
      acc = wmma_bf16(a, bb, acc);
    }
#pragma unroll
    for (int r = 0; r < 8; ++r) {
      int row = m0 + half * 8 + r;
      float v = acc[r] > 0.f ? acc[r] : 0.f;
      out_g[(size_t)(b * C_ + row) * HW_ + hw0 + ln] =
          v + arm_g[(size_t)(b * C_ + row) * HW_ + hw0 + ln];
    }
  }
}

// ---------------------------------------------------------------------------
extern "C" void kernel_launch(void* const* d_in, const int* in_sizes, int n_in,
                              void* d_out, int out_size, void* d_ws, size_t ws_size,
                              hipStream_t stream) {
  (void)in_sizes; (void)n_in; (void)out_size; (void)ws_size;
  const float* feat_l   = (const float*)d_in[0];
  const float* feat_s   = (const float*)d_in[1];
  const float* fsm_w    = (const float*)d_in[2];
  const float* offset_w = (const float*)d_in[3];
  const float* com_w    = (const float*)d_in[4];
  const float* com_b    = (const float*)d_in[5];
  const float* dcn_w    = (const float*)d_in[6];
  const float* dcn_b    = (const float*)d_in[7];
  const float* cat_w    = (const float*)d_in[8];
  float* out = (float*)d_out;

  // Workspace layout (256B aligned):
  //   om      fp32  28,311,552 B
  //   arm     fp32  16,777,216 B
  //   off     bf16   8,388,608 B
  //   weights bf16    ~759,808 B          (total ~54.3 MB)
  char* ws = (char*)d_ws;
  size_t o = 0;
  float*  om_g   = (float*)(ws + o);  o += (size_t)B_ * OMC_ * HW_ * 4;
  float*  arm_g  = (float*)(ws + o);  o += (size_t)B_ * C_ * HW_ * 4;
  bf16_t* off_bf = (bf16_t*)(ws + o); o += (size_t)B_ * C_ * HW_ * 2;
  bf16_t* fsm_b  = (bf16_t*)(ws + o); o += (size_t)C_ * C_ * 2;
  bf16_t* offw_b = (bf16_t*)(ws + o); o += (size_t)C_ * 2 * C_ * 2;
  bf16_t* comt_b = (bf16_t*)(ws + o); o += (size_t)9 * OMC_ * C_ * 2;
  bf16_t* dcnw_b = (bf16_t*)(ws + o); o += (size_t)COUT_ * KDCN_ * 2;
  bf16_t* catw_b = (bf16_t*)(ws + o); o += (size_t)C_ * COUT_ * 2;

  const int ntiles = B_ * HW_ / 16;  // 2048 pixel tiles of 16

  k0_cvt_weights<<<128, 256, 0, stream>>>(fsm_w, offset_w, com_w, dcn_w, cat_w,
                                          fsm_b, offw_b, comt_b, dcnw_b, catw_b);
  k1_fsm_offset<<<ntiles, 256, 0, stream>>>(feat_l, feat_s, fsm_b, offw_b,
                                            arm_g, off_bf);
  k2_com_conv<<<dim3(ntiles, 2), 224, 0, stream>>>(off_bf, comt_b, com_b, om_g);
  k3_dcn<<<ntiles, 128, 0, stream>>>(feat_s, om_g, dcnw_b, dcn_b, catw_b,
                                     arm_g, out);
}
